// LabelClassifier_1554778161814
// MI455X (gfx1250) — matrix operand order repaired and checked by reading, not compiled, not run on hardware
//
#include <hip/hip_runtime.h>
#include <math.h>

typedef __attribute__((ext_vector_type(16))) __bf16 v16bf;
typedef __attribute__((ext_vector_type(8)))  __bf16 v8bf;
typedef __attribute__((ext_vector_type(8)))  float  v8f;

#define S_DIM 2048
#define B_DIM 16
#define H_DIM 512
#define L_DIM 4096
#define LAB   256
#define V_DIM 32000
#define SPLITS 8
#define COLS_PER_SLICE (V_DIM / SPLITS)        /* 4000 */
#define TILES_PER_SLICE (COLS_PER_SLICE / 16)  /* 250  */

#if defined(__has_builtin)
#if __has_builtin(__builtin_amdgcn_global_load_async_to_lds_b128)
#define HAVE_ASYNC_LDS 1
#endif
#if __has_builtin(__builtin_amdgcn_s_wait_asynccnt)
#define HAVE_WAIT_ASYNC_BUILTIN 1
#endif
#endif

// Builtin wants AS(1)/AS(3) pointers to 128-bit int vectors (per hipcc diagnostic).
typedef __attribute__((__vector_size__(16))) int v4i_t;
typedef __attribute__((address_space(1))) v4i_t g_v4i;
typedef __attribute__((address_space(3))) v4i_t l_v4i;

static __device__ __forceinline__ g_v4i* to_global_v4(const void* p) {
    return (g_v4i*)(uintptr_t)p;
}
static __device__ __forceinline__ l_v4i* to_lds_v4(void* p) {
    return (l_v4i*)(unsigned)(uintptr_t)p;      // generic->LDS: low 32 bits are LDS offset
}

static __device__ __forceinline__ void wait_async0() {
#if defined(HAVE_WAIT_ASYNC_BUILTIN)
    __builtin_amdgcn_s_wait_asynccnt(0);
#else
    asm volatile("s_wait_asynccnt 0x0" ::: "memory");
#endif
}

static __device__ __forceinline__ unsigned short f2bf(float f) {
    unsigned u = __float_as_uint(f);
    u += 0x7FFFu + ((u >> 16) & 1u);           // round-to-nearest-even
    return (unsigned short)(u >> 16);
}
static __device__ __forceinline__ float bf2f(unsigned short h) {
    return __uint_as_float((unsigned)h << 16);
}

union FragBF { v16bf v; v8bf h[2]; };

// A fragment: 16x32 bf16 from row-major [M][ld].
static __device__ __forceinline__ v16bf loadFragA(const unsigned short* A, int ld,
                                                  int r0, int k0, int lane) {
    const unsigned short* p = A + (size_t)(r0 + (lane & 15)) * ld + k0 + ((lane >> 4) << 4);
    FragBF f;
    f.h[0] = *(const v8bf*)p;
    f.h[1] = *(const v8bf*)(p + 8);
    return f.v;
}

// B fragment: 32x16 bf16 from column-transposed weights [N][ld] (K contiguous).
static __device__ __forceinline__ v16bf loadFragB(const unsigned short* BT, int ld,
                                                  int c0, int k0, int lane) {
    const unsigned short* p = BT + (size_t)(c0 + (lane & 15)) * ld + k0 + ((lane >> 4) << 4);
    FragBF f;
    f.h[0] = *(const v8bf*)p;
    f.h[1] = *(const v8bf*)(p + 8);
    return f.v;
}

// ---------- weight transpose + f32->bf16 : src[K][N] -> dst[N][K] ----------
__global__ void transpose_bf16_kernel(const float* __restrict__ src,
                                      unsigned short* __restrict__ dst,
                                      int K, int N) {
    size_t total = (size_t)K * N;
    for (size_t i = (size_t)blockIdx.x * blockDim.x + threadIdx.x; i < total;
         i += (size_t)gridDim.x * blockDim.x) {
        int k = (int)(i / N);
        int n = (int)(i % N);
        dst[(size_t)n * K + k] = f2bf(src[i]);
    }
}

// ---------- span embedding gather: [left | mean | right] -> bf16 [L][1536] ----------
__global__ void span_kernel(const float* __restrict__ hidden,
                            const int* __restrict__ begins,
                            const int* __restrict__ ends,
                            const int* __restrict__ bids,
                            unsigned short* __restrict__ spanEmb) {
    int l = blockIdx.x;
    int b = begins[l], e = ends[l], bid = bids[l];
    float invLen = 1.0f / (float)(e - b);
    size_t row = (size_t)l * (3 * H_DIM);
    for (int c = threadIdx.x; c < H_DIM; c += blockDim.x) {
        float left  = hidden[((size_t)(b - 1) * B_DIM + bid) * H_DIM + c];
        float right = hidden[((size_t)e       * B_DIM + bid) * H_DIM + c];
        float acc = 0.f;
        for (int p = b; p < e; ++p)
            acc += hidden[((size_t)p * B_DIM + bid) * H_DIM + c];
        spanEmb[row + c]             = f2bf(left);
        spanEmb[row + H_DIM + c]     = f2bf(acc * invLen);
        spanEmb[row + 2 * H_DIM + c] = f2bf(right);
    }
}

// ---------- generic WMMA GEMM: C[M][N] = act(A[M][K] @ BT[N][K]^T + bias) ----------
__global__ __launch_bounds__(256)
void mlp_gemm_kernel(const unsigned short* __restrict__ A,
                     const unsigned short* __restrict__ BT,
                     const float* __restrict__ bias,
                     unsigned short* __restrict__ C,
                     int N, int K, int sigmoidFlag) {
    int wave = blockIdx.x * (blockDim.x >> 5) + (threadIdx.x >> 5);
    int lane = threadIdx.x & 31;
    int tilesN = N >> 4;
    int c0 = (wave % tilesN) << 4;
    int r0 = (wave / tilesN) << 4;

    v8f acc = {};
    for (int k0 = 0; k0 < K; k0 += 32) {
        v16bf a = loadFragA(A,  K, r0, k0, lane);
        v16bf b = loadFragB(BT, K, c0, k0, lane);
        acc = __builtin_amdgcn_wmma_f32_16x16x32_bf16(false, a, false, b,
                                                      (short)0, acc, false, false);
    }
    int col   = c0 + (lane & 15);
    int rbase = r0 + ((lane >> 4) << 3);
    float bc = bias[col];
#pragma unroll
    for (int j = 0; j < 8; ++j) {
        float x = acc[j] + bc;
        if (sigmoidFlag) x = 1.0f / (1.0f + __expf(-x));
        C[(size_t)(rbase + j) * N + col] = f2bf(x);
    }
}

// ---------- classifier GEMM fused with streaming logsumexp ----------
// Block = 8 waves, ALL on the same column slice, 8 consecutive row tiles.
// B tiles staged once per block in LDS (double-buffered, async-to-LDS).
__global__ __launch_bounds__(256)
void logits_lse_kernel(const unsigned short* __restrict__ feat, // [L][LAB] bf16
                       const unsigned short* __restrict__ WoT,  // [V][LAB] bf16
                       const float* __restrict__ bo,
                       float* __restrict__ mOut,
                       float* __restrict__ sOut) {
    __shared__ unsigned short ldsB[2][16 * LAB];   // 2 x 8KB

    int lane   = threadIdx.x & 31;
    int wv     = threadIdx.x >> 5;                 // 0..7
    int slice  = blockIdx.x & (SPLITS - 1);
    int rowBlk = blockIdx.x >> 3;                  // 0..31
    int r0     = (rowBlk * 8 + wv) * 16;
    int cBeg   = slice * COLS_PER_SLICE;

    // Hoist all A fragments (16 rows x K=256) into registers: 8 frags = 64 VGPRs.
    v16bf a[8];
#pragma unroll
    for (int kk = 0; kk < 8; ++kk) a[kk] = loadFragA(feat, LAB, r0, kk * 32, lane);

    float m[8], s[8], base[8];
#pragma unroll
    for (int j = 0; j < 8; ++j) { m[j] = -__builtin_inff(); s[j] = 0.f; base[j] = 0.f; }

    // stage helper: copy one 8KB B tile (16 cols x 256 K, row-major) into LDS buffer
    auto stage = [&](int t, int buf) {
        const unsigned short* g = WoT + (size_t)(cBeg + t * 16) * LAB;
        unsigned short* d = &ldsB[buf][0];
        int e0 = threadIdx.x * 8;               // two 16B chunks per thread
#if defined(HAVE_ASYNC_LDS)
        __builtin_amdgcn_global_load_async_to_lds_b128(to_global_v4(g + e0),
                                                       to_lds_v4(d + e0), 0, 0);
        __builtin_amdgcn_global_load_async_to_lds_b128(to_global_v4(g + e0 + 2048),
                                                       to_lds_v4(d + e0 + 2048), 0, 0);
#else
        *(v8bf*)(d + e0)        = *(const v8bf*)(g + e0);
        *(v8bf*)(d + e0 + 2048) = *(const v8bf*)(g + e0 + 2048);
#endif
    };

    stage(0, 0);
#if defined(HAVE_ASYNC_LDS)
    wait_async0();
#endif
    __syncthreads();

    int cur = 0;
    for (int t = 0; t < TILES_PER_SLICE; ++t) {
        if (t + 1 < TILES_PER_SLICE) stage(t + 1, cur ^ 1);

        // accumulator initialized with the output bias (folds the bias add)
        float bc = bo[cBeg + t * 16 + (lane & 15)];
        v8f acc = {bc, bc, bc, bc, bc, bc, bc, bc};
        const unsigned short* bbase = &ldsB[cur][0] + (lane & 15) * LAB + ((lane >> 4) << 4);
#pragma unroll
        for (int kk = 0; kk < 8; ++kk) {
            const unsigned short* bp = bbase + kk * 32;
            FragBF f;
            f.h[0] = *(const v8bf*)bp;
            f.h[1] = *(const v8bf*)(bp + 8);
            acc = __builtin_amdgcn_wmma_f32_16x16x32_bf16(false, a[kk], false, f.v,
                                                          (short)0, acc, false, false);
        }
        // cheap streaming logsumexp: running max + sum of exp(x - base)
#pragma unroll
        for (int j = 0; j < 8; ++j) {
            float x = acc[j];
            m[j] = fmaxf(m[j], x);
            s[j] += __expf(x - base[j]);
        }
        // periodically move base up to the running max (overflow safety)
        if ((t & 31) == 7) {
#pragma unroll
            for (int j = 0; j < 8; ++j) {
                s[j] *= __expf(base[j] - m[j]);
                base[j] = m[j];
            }
        }

#if defined(HAVE_ASYNC_LDS)
        wait_async0();
#endif
        __syncthreads();
        cur ^= 1;
    }

    // final fold so (base, s) is a canonical (max, sumexp) pair
#pragma unroll
    for (int j = 0; j < 8; ++j) {
        s[j] *= __expf(base[j] - m[j]);
        base[j] = m[j];
    }

    // merge across the 16 lanes sharing each row half (xor 1,2,4,8 keeps halves apart)
#pragma unroll
    for (int off = 1; off < 16; off <<= 1) {
#pragma unroll
        for (int j = 0; j < 8; ++j) {
            float m2 = __shfl_xor(base[j], off);
            float s2 = __shfl_xor(s[j], off);
            float nm = fmaxf(base[j], m2);
            s[j] = s[j] * __expf(base[j] - nm) + s2 * __expf(m2 - nm);
            base[j] = nm;
        }
    }
    if ((lane & 15) == 0) {
        int rbase = r0 + ((lane >> 4) << 3);
#pragma unroll
        for (int j = 0; j < 8; ++j) {
            size_t idx = (size_t)(rbase + j) * SPLITS + slice;
            mOut[idx] = base[j];
            sOut[idx] = s[j];
        }
    }
}

// ---------- deterministic final merge + tag-logit dot + NLL reduction ----------
__global__ __launch_bounds__(256)
void final_reduce_kernel(const float* __restrict__ mA,
                         const float* __restrict__ sA,
                         const unsigned short* __restrict__ feat, // [L][LAB] bf16
                         const unsigned short* __restrict__ WoT,  // [V][LAB] bf16
                         const float* __restrict__ bo,
                         const int* __restrict__ tags,
                         float* __restrict__ out) {
    __shared__ float red[256];
    float localSum = 0.f;
    for (int row = threadIdx.x; row < L_DIM; row += blockDim.x) {
        float m = -__builtin_inff();
        for (int k = 0; k < SPLITS; ++k) m = fmaxf(m, mA[row * SPLITS + k]);
        float ss = 0.f;
        for (int k = 0; k < SPLITS; ++k)
            ss += sA[row * SPLITS + k] * __expf(mA[row * SPLITS + k] - m);
        // tag logit = feat[row] . WoT[tag] + bo[tag]
        int tag = tags[row];
        const unsigned short* fr = feat + (size_t)row * LAB;
        const unsigned short* wr = WoT + (size_t)tag * LAB;
        float tv = bo[tag];
        for (int k = 0; k < LAB; ++k) tv += bf2f(fr[k]) * bf2f(wr[k]);
        localSum += (m + __logf(ss)) - tv;     // -logp[tag]
    }
    red[threadIdx.x] = localSum;
    __syncthreads();
    for (int off = 128; off > 0; off >>= 1) {
        if ((int)threadIdx.x < off) red[threadIdx.x] += red[threadIdx.x + off];
        __syncthreads();
    }
    if (threadIdx.x == 0) out[0] = red[0] / ((float)L_DIM + 1e-5f);
}

extern "C" void kernel_launch(void* const* d_in, const int* in_sizes, int n_in,
                              void* d_out, int out_size, void* d_ws, size_t ws_size,
                              hipStream_t stream) {
    (void)in_sizes; (void)n_in; (void)out_size; (void)ws_size;
    const float* hidden = (const float*)d_in[0];
    const int*   begins = (const int*)d_in[1];
    const int*   ends   = (const int*)d_in[2];
    const int*   bids   = (const int*)d_in[3];
    const int*   tags   = (const int*)d_in[4];
    const float* W1     = (const float*)d_in[5];
    const float* b1     = (const float*)d_in[6];
    const float* W2     = (const float*)d_in[7];
    const float* b2     = (const float*)d_in[8];
    const float* Wo     = (const float*)d_in[9];
    const float* bo     = (const float*)d_in[10];

    char* ws = (char*)d_ws;
    size_t off = 0;
    auto carve = [&](size_t bytes) -> char* {
        off = (off + 255) & ~(size_t)255;
        char* p = ws + off;
        off += bytes;
        return p;
    };
    unsigned short* spanEmb = (unsigned short*)carve((size_t)L_DIM * 3 * H_DIM * 2);
    unsigned short* W1T     = (unsigned short*)carve((size_t)3 * H_DIM * LAB * 2);
    unsigned short* h1      = (unsigned short*)carve((size_t)L_DIM * LAB * 2);
    unsigned short* W2T     = (unsigned short*)carve((size_t)LAB * LAB * 2);
    unsigned short* feat    = (unsigned short*)carve((size_t)L_DIM * LAB * 2);
    unsigned short* WoT     = (unsigned short*)carve((size_t)V_DIM * LAB * 2);
    float* mA = (float*)carve((size_t)L_DIM * SPLITS * 4);
    float* sA = (float*)carve((size_t)L_DIM * SPLITS * 4);

    transpose_bf16_kernel<<<256,  256, 0, stream>>>(W1, W1T, 3 * H_DIM, LAB);
    transpose_bf16_kernel<<<64,   256, 0, stream>>>(W2, W2T, LAB, LAB);
    transpose_bf16_kernel<<<2048, 256, 0, stream>>>(Wo, WoT, LAB, V_DIM);

    span_kernel<<<L_DIM, 256, 0, stream>>>(hidden, begins, ends, bids, spanEmb);

    // (L/16)*(LAB/16) = 4096 tiles, 8 waves/block -> 512 blocks
    mlp_gemm_kernel<<<512, 256, 0, stream>>>(spanEmb, W1T, b1, h1, LAB, 3 * H_DIM, 1);
    mlp_gemm_kernel<<<512, 256, 0, stream>>>(h1,      W2T, b2, feat, LAB, LAB,     0);

    // 256 blocks: blockIdx = rowBlk*8 + slice ; each block = 8 waves on one slice
    logits_lse_kernel<<<256, 256, 0, stream>>>(feat, WoT, bo, mA, sA);

    final_reduce_kernel<<<1, 256, 0, stream>>>(mA, sA, feat, WoT, bo, tags, (float*)d_out);
}
